// Encoder_3908420239370
// MI455X (gfx1250) — matrix operand order repaired and checked by reading, not compiled, not run on hardware
//
#include <hip/hip_runtime.h>

typedef __attribute__((ext_vector_type(16))) __bf16 v16bf;
typedef __attribute__((ext_vector_type(8)))  float  v8f;
typedef __attribute__((ext_vector_type(8)))  unsigned int v8u;
typedef __attribute__((ext_vector_type(4)))  unsigned int v4u;
typedef __attribute__((ext_vector_type(8)))  int v8i;
typedef __attribute__((ext_vector_type(4)))  int v4i;

#if __has_builtin(__builtin_amdgcn_tensor_load_to_lds) && \
    __has_builtin(__builtin_amdgcn_s_wait_tensorcnt)
#define USE_TDM 1
#endif

// ---------- bf16 helpers (round-to-nearest-even) ----------
__device__ __forceinline__ unsigned short f2bf_us(float f) {
  unsigned u = __builtin_bit_cast(unsigned, f);
  u += 0x7FFFu + ((u >> 16) & 1u);
  return (unsigned short)(u >> 16);
}

// ---------- cross-lane reductions (VALU permlane, no LDS) ----------
#if __has_builtin(__builtin_amdgcn_permlane16) && \
    __has_builtin(__builtin_amdgcn_permlanex16)
__device__ __forceinline__ float plane16(float v, unsigned lo, unsigned hi) {
  unsigned u = __builtin_bit_cast(unsigned, v);
  u = __builtin_amdgcn_permlane16(u, u, lo, hi, false, false);
  return __builtin_bit_cast(float, u);
}
__device__ __forceinline__ float planex16(float v) {  // lane ^ 16
  unsigned u = __builtin_bit_cast(unsigned, v);
  u = __builtin_amdgcn_permlanex16(u, u, 0x76543210u, 0xFEDCBA98u, false, false);
  return __builtin_bit_cast(float, u);
}
__device__ __forceinline__ float redmax16(float v) {
  v = fmaxf(v, plane16(v, 0x67452301u, 0xEFCDAB89u));
  v = fmaxf(v, plane16(v, 0x54761032u, 0xDCFE98BAu));
  v = fmaxf(v, plane16(v, 0x32107654u, 0xBA98FEDCu));
  v = fmaxf(v, plane16(v, 0xFEDCBA98u, 0x76543210u));
  return v;
}
__device__ __forceinline__ float redsum16(float v) {
  v += plane16(v, 0x67452301u, 0xEFCDAB89u);
  v += plane16(v, 0x54761032u, 0xDCFE98BAu);
  v += plane16(v, 0x32107654u, 0xBA98FEDCu);
  v += plane16(v, 0xFEDCBA98u, 0x76543210u);
  return v;
}
__device__ __forceinline__ float redsum32(float v) {
  v = redsum16(v);
  v += planex16(v);
  return v;
}
#else
__device__ __forceinline__ float redmax16(float v) {
  for (int m = 8; m >= 1; m >>= 1) v = fmaxf(v, __shfl_xor(v, m, 32));
  return v;
}
__device__ __forceinline__ float redsum16(float v) {
  for (int m = 8; m >= 1; m >>= 1) v += __shfl_xor(v, m, 32);
  return v;
}
__device__ __forceinline__ float redsum32(float v) {
  for (int m = 16; m >= 1; m >>= 1) v += __shfl_xor(v, m, 32);
  return v;
}
#endif

// ---------- TDM: async 2D bf16 tile, global -> LDS (padded rows) ----------
// pad_interval code: insert padding every (2<<code) DWORDs stored.
// pad_amount code: padding = (code+1) DWORDs.
__device__ __forceinline__ void tdm_load_2d(unsigned lds_off, const void* gptr,
                                            unsigned tensor_d0,
                                            unsigned tensor_d1,
                                            unsigned tile_d0, unsigned tile_d1,
                                            unsigned long long stride0,
                                            unsigned pad_i, unsigned pad_a) {
#ifdef USE_TDM
  unsigned long long ga = (unsigned long long)gptr;
  v4u g0;
  g0[0] = 1u;                                                // count=1, user
  g0[1] = lds_off;                                           // lds_addr
  g0[2] = (unsigned)(ga & 0xFFFFFFFFu);                      // global[31:0]
  g0[3] = (unsigned)((ga >> 32) & 0x01FFFFFFu) | (2u << 30); // type=2 (image)
  v8i g1;
  g1[0] = (int)((1u << 16) |                  // data_size = 2 bytes
                (1u << 20) |                  // pad_enable
                (pad_i << 22) | (pad_a << 25));
  g1[1] = (int)((tensor_d0 & 0xFFFFu) << 16);                // abar=0 | td0.lo
  g1[2] = (int)(((tensor_d0 >> 16) & 0xFFFFu) | ((tensor_d1 & 0xFFFFu) << 16));
  g1[3] = (int)(((tensor_d1 >> 16) & 0xFFFFu) | ((tile_d0 & 0xFFFFu) << 16));
  g1[4] = (int)(tile_d1 & 0xFFFFu);                          // tile_dim2 = 0
  g1[5] = (int)(unsigned)(stride0 & 0xFFFFFFFFull);
  g1[6] = (int)(unsigned)((stride0 >> 32) & 0xFFFFull);      // s1.lo16 = 0
  g1[7] = 0;
  v4i z4 = {0, 0, 0, 0};
  v8i z8 = {0, 0, 0, 0, 0, 0, 0, 0};
  __builtin_amdgcn_tensor_load_to_lds(g0, g1, z4, z4, z8, 0);
#else
  (void)lds_off; (void)gptr; (void)tensor_d0; (void)tensor_d1;
  (void)tile_d0; (void)tile_d1; (void)stride0; (void)pad_i; (void)pad_a;
#endif
}

// Load a 16-element bf16 fragment as two 16B chunks (pointers in ushort units).
__device__ __forceinline__ v16bf ld16(const unsigned short* p0,
                                      const unsigned short* p1) {
  uint4 a = *(const uint4*)p0;
  uint4 b = *(const uint4*)p1;
  v8u u;
  u[0] = a.x; u[1] = a.y; u[2] = a.z; u[3] = a.w;
  u[4] = b.x; u[5] = b.y; u[6] = b.z; u[7] = b.w;
  return __builtin_bit_cast(v16bf, u);
}
// A-fragment (16x32): row = lane&15; two contiguous 8-ushort runs at
// k = 8*hh and k = 16 + 8*hh  (hh = lane>>4)
__device__ __forceinline__ v16bf load_a_frag(const unsigned short* row, int hh) {
  return ld16(row + 8 * hh, row + 16 + 8 * hh);
}
// B-fragment (32x16) from row-major B[k][n]: n = lane&15, elements over k at
// k = 8*hh + s (s<8) and k = 16 + 8*hh + (s-8) (s>=8)  (hh = lane>>4)
__device__ __forceinline__ v16bf load_b_frag(const unsigned short* base, int ld,
                                             int lane) {
  const unsigned short* p = base + (lane & 15);
  const int hb = lane >> 4;
  v16bf r;
#pragma unroll
  for (int s = 0; s < 16; ++s) {
    int k = (s < 8) ? (8 * hb + s) : (16 + 8 * hb + (s - 8));
    r[s] = __builtin_bit_cast(__bf16, p[k * ld]);
  }
  return r;
}

// ---------- LayerNorm: f32 in -> bf16 out, one row per wave ----------
template <int C>
__global__ __launch_bounds__(256) void ln_bf16(const float* __restrict__ x,
                                               const float* __restrict__ g,
                                               const float* __restrict__ bta,
                                               unsigned short* __restrict__ out) {
  constexpr int PER = C / 32;
  const int wv = threadIdx.x >> 5, lane = threadIdx.x & 31;
  const long row = (long)blockIdx.x * 8 + wv;
  const float* xr = x + row * C;
  float v[PER];
#pragma unroll
  for (int i = 0; i < PER; i += 4) {
    float4 t = *(const float4*)(xr + lane * PER + i);
    v[i] = t.x; v[i + 1] = t.y; v[i + 2] = t.z; v[i + 3] = t.w;
  }
  float s = 0.f;
#pragma unroll
  for (int i = 0; i < PER; ++i) s += v[i];
  s = redsum32(s);
  const float mean = s * (1.0f / C);
  float vs = 0.f;
#pragma unroll
  for (int i = 0; i < PER; ++i) { float d = v[i] - mean; vs += d * d; }
  vs = redsum32(vs);
  const float rstd = rsqrtf(vs * (1.0f / C) + 1e-5f);
  unsigned o[PER / 2];
#pragma unroll
  for (int i = 0; i < PER; i += 2) {
    int c = lane * PER + i;
    unsigned short a = f2bf_us((v[i] - mean) * rstd * g[c] + bta[c]);
    unsigned short b = f2bf_us((v[i + 1] - mean) * rstd * g[c + 1] + bta[c + 1]);
    o[i / 2] = (unsigned)a | ((unsigned)b << 16);
  }
  if constexpr (PER == 4) {
    *(uint2*)&out[row * C + lane * PER] = make_uint2(o[0], o[1]);
  } else {
    *(uint4*)&out[row * C + lane * PER] = make_uint4(o[0], o[1], o[2], o[3]);
  }
}

// ---------- GEMM epilogues ----------
constexpr int EPI_BF16 = 0;       // out bf16 row-major = acc + bias
constexpr int EPI_GELU_BF16 = 1;  // out bf16 = gelu(acc + bias)
constexpr int EPI_RES_F32 = 2;    // out f32 (strided/col-offset) = acc+bias+res
constexpr int EPI_F32_GUARD = 3;  // out f32, store only col < N
constexpr int EPI_BF16_T = 4;     // out bf16 transposed per batch [B][N][Lb]

template <int EPI>
__device__ __forceinline__ void store_tile(const v8f& acc, float bv,
                                           long rowBase, int rowT, int hh,
                                           int l15, int col, int N,
                                           const float* __restrict__ res,
                                           int res_stride,
                                           void* __restrict__ out,
                                           int out_rstride, int out_coff) {
  if constexpr (EPI == EPI_BF16_T) {
    long row0 = rowBase + rowT * 16 + 8 * hh;  // multiple of 8
    int Lb = out_rstride;                      // per-batch row count
    long b = row0 / Lb, rl = row0 % Lb;
    unsigned o[4];
#pragma unroll
    for (int i = 0; i < 4; ++i) {
      unsigned short a0 = f2bf_us(acc[2 * i] + bv);
      unsigned short a1 = f2bf_us(acc[2 * i + 1] + bv);
      o[i] = (unsigned)a0 | ((unsigned)a1 << 16);
    }
    *(uint4*)&((unsigned short*)out)[((size_t)b * N + col) * Lb + rl] =
        make_uint4(o[0], o[1], o[2], o[3]);
  } else {
#pragma unroll
    for (int r = 0; r < 8; ++r) {
      long row = rowBase + rowT * 16 + r + 8 * hh;
      float v = acc[r] + bv;
      if (EPI == EPI_GELU_BF16) v = 0.5f * v * (1.f + erff(v * 0.70710678118f));
      if (EPI == EPI_BF16 || EPI == EPI_GELU_BF16) {
        ((unsigned short*)out)[row * (long)out_rstride + col] = f2bf_us(v);
      } else if (EPI == EPI_RES_F32) {
        v += res[row * (long)res_stride + col];
        ((float*)out)[row * (long)out_rstride + out_coff + col] = v;
      } else {  // EPI_F32_GUARD
        if (col < N) ((float*)out)[row * (long)out_rstride + col] = v;
      }
    }
  }
}

// stage one 64xNB W chunk (f32 -> bf16) into LDS (row stride NB+8)
template <int NB>
__device__ __forceinline__ void stage_w(const float* __restrict__ W, int K,
                                        int N, int kc, int colBase, int tid,
                                        unsigned short* Bs) {
  constexpr int KC = 64;
  for (int i = tid; i < KC * (NB / 4); i += 256) {
    int r = i / (NB / 4);
    int c = (i % (NB / 4)) * 4;
    int gc = colBase + c;
    const float* wp = W + (long)(kc + r) * N + gc;
    float w0, w1, w2, w3;
    if (gc + 3 < N) {
      float4 fv = *(const float4*)wp;
      w0 = fv.x; w1 = fv.y; w2 = fv.z; w3 = fv.w;
    } else {
      w0 = (gc + 0 < N) ? wp[0] : 0.f;
      w1 = (gc + 1 < N) ? wp[1] : 0.f;
      w2 = (gc + 2 < N) ? wp[2] : 0.f;
      w3 = (gc + 3 < N) ? wp[3] : 0.f;
    }
    unsigned lo = (unsigned)f2bf_us(w0) | ((unsigned)f2bf_us(w1) << 16);
    unsigned hi = (unsigned)f2bf_us(w2) | ((unsigned)f2bf_us(w3) << 16);
    *(uint2*)&Bs[r * (NB + 8) + c] = make_uint2(lo, hi);
  }
}

// fallback A-chunk staging (64 rows x 64 bf16, row stride 72)
__device__ __forceinline__ void stage_a(const unsigned short* __restrict__ A,
                                        long rowBase, long K, int kc, int tid,
                                        unsigned short* As) {
#ifndef USE_TDM
  for (int i = tid; i < 64 * 8; i += 256) {
    int r = i / 8;
    int c = (i % 8) * 8;
    *(uint4*)&As[r * 72 + c] = *(const uint4*)(A + (rowBase + r) * K + kc + c);
  }
#else
  (void)A; (void)rowBase; (void)K; (void)kc; (void)tid; (void)As;
#endif
}

// ---------- Tiled GEMM: bf16 A [R,K] x f32 W [K,N] ----------
template <int EPI, int NB>
__global__ __launch_bounds__(256) void gemm_bf16(
    const unsigned short* __restrict__ A, const float* __restrict__ W,
    const float* __restrict__ bias, const float* __restrict__ res,
    void* __restrict__ out, int K, int N, int res_stride, int out_rstride,
    int out_coff) {
  constexpr int KC = 64;
  constexpr int TPW = NB / 32;
  extern __shared__ unsigned short smem[];
  unsigned short* As = smem;               // [64][72], 9216 B at LDS offset 0
  unsigned short* Bs = smem + 64 * 72;     // [64][NB+8]
  const int tid = threadIdx.x, wv = tid >> 5, lane = tid & 31;
  const int hh = lane >> 4, l15 = lane & 15;
  const int rowT = wv & 3, colTb = wv >> 2;
  const long rowBase = (long)blockIdx.x * 64;
  const int colBase = blockIdx.y * NB;

  v8f acc[TPW];
#pragma unroll
  for (int t = 0; t < TPW; ++t)
#pragma unroll
    for (int r = 0; r < 8; ++r) acc[t][r] = 0.f;

  for (int kc = 0; kc < K; kc += KC) {
    __syncthreads();
#ifdef USE_TDM
    if (wv == 0)  // async A tile: 64 rows x 64 bf16, 128B rows + 16B pad
      tdm_load_2d(0u, A + rowBase * K + kc, K, 1u << 20, 64, 64,
                  (unsigned long long)K, 4u, 3u);
#else
    stage_a(A, rowBase, K, kc, tid, As);
#endif
    stage_w<NB>(W, K, N, kc, colBase, tid, Bs);
#ifdef USE_TDM
    if (wv == 0) __builtin_amdgcn_s_wait_tensorcnt(0);
#endif
    __syncthreads();
#pragma unroll
    for (int k0 = 0; k0 < KC; k0 += 32) {
      v16bf a = load_a_frag(&As[(rowT * 16 + l15) * 72 + k0], hh);
#pragma unroll
      for (int t = 0; t < TPW; ++t) {
        int colT = colTb + 2 * t;
        v16bf b = load_b_frag(&Bs[k0 * (NB + 8) + colT * 16], NB + 8, lane);
        acc[t] = __builtin_amdgcn_wmma_f32_16x16x32_bf16(
            false, a, false, b, (short)0, acc[t], false, false);
      }
    }
  }
#pragma unroll
  for (int t = 0; t < TPW; ++t) {
    int colT = colTb + 2 * t;
    int col = colBase + colT * 16 + l15;
    float bv = (col < N) ? bias[col] : 0.f;
    store_tile<EPI>(acc[t], bv, rowBase, rowT, hh, l15, col, N, res, res_stride,
                    out, out_rstride, out_coff);
  }
}

// ---------- Dual GEMM: one A tile, two weight matrices / outputs ----------
template <int EPI1, int EPI2, int NB>
__global__ __launch_bounds__(256) void gemm_dual(
    const unsigned short* __restrict__ A, const float* __restrict__ W1,
    const float* __restrict__ b1, void* __restrict__ out1, int rstride1,
    const float* __restrict__ W2, const float* __restrict__ b2,
    void* __restrict__ out2, int rstride2, int K, int N) {
  constexpr int KC = 64;
  constexpr int TPW = NB / 32;
  extern __shared__ unsigned short smem[];
  unsigned short* As = smem;                           // [64][72]
  unsigned short* Bs1 = smem + 64 * 72;                // [64][NB+8]
  unsigned short* Bs2 = Bs1 + 64 * (NB + 8);
  const int tid = threadIdx.x, wv = tid >> 5, lane = tid & 31;
  const int hh = lane >> 4, l15 = lane & 15;
  const int rowT = wv & 3, colTb = wv >> 2;
  const long rowBase = (long)blockIdx.x * 64;

  v8f acc1[TPW], acc2[TPW];
#pragma unroll
  for (int t = 0; t < TPW; ++t)
#pragma unroll
    for (int r = 0; r < 8; ++r) { acc1[t][r] = 0.f; acc2[t][r] = 0.f; }

  for (int kc = 0; kc < K; kc += KC) {
    __syncthreads();
#ifdef USE_TDM
    if (wv == 0)
      tdm_load_2d(0u, A + rowBase * K + kc, K, 1u << 20, 64, 64,
                  (unsigned long long)K, 4u, 3u);
#else
    stage_a(A, rowBase, K, kc, tid, As);
#endif
    stage_w<NB>(W1, K, N, kc, 0, tid, Bs1);
    stage_w<NB>(W2, K, N, kc, 0, tid, Bs2);
#ifdef USE_TDM
    if (wv == 0) __builtin_amdgcn_s_wait_tensorcnt(0);
#endif
    __syncthreads();
#pragma unroll
    for (int k0 = 0; k0 < KC; k0 += 32) {
      v16bf a = load_a_frag(&As[(rowT * 16 + l15) * 72 + k0], hh);
#pragma unroll
      for (int t = 0; t < TPW; ++t) {
        int colT = colTb + 2 * t;
        v16bf bb1 = load_b_frag(&Bs1[k0 * (NB + 8) + colT * 16], NB + 8, lane);
        acc1[t] = __builtin_amdgcn_wmma_f32_16x16x32_bf16(
            false, a, false, bb1, (short)0, acc1[t], false, false);
        v16bf bb2 = load_b_frag(&Bs2[k0 * (NB + 8) + colT * 16], NB + 8, lane);
        acc2[t] = __builtin_amdgcn_wmma_f32_16x16x32_bf16(
            false, a, false, bb2, (short)0, acc2[t], false, false);
      }
    }
  }
#pragma unroll
  for (int t = 0; t < TPW; ++t) {
    int colT = colTb + 2 * t;
    int col = colT * 16 + l15;
    store_tile<EPI1>(acc1[t], b1[col], rowBase, rowT, hh, l15, col, N, nullptr,
                     0, out1, rstride1, 0);
    store_tile<EPI2>(acc2[t], b2[col], rowBase, rowT, hh, l15, col, N, nullptr,
                     0, out2, rstride2, 0);
  }
}

// ---------- Flash attention: Q=k2 (row-major), K=q1 (pre-transposed) ----------
// dynamic LDS layout (bytes):
//   Kts [128][72]  at     0  (18432)
//   V1s [ 64][136] at 18432  (17408)
//   V2s [ 64][136] at 35840  (17408)
//   Ps  [8][16][40] at 53248 (10240)   -> total 63488
__global__ __launch_bounds__(256) void flash_attn(
    const unsigned short* __restrict__ Q,    // [B][L][128] bf16
    const unsigned short* __restrict__ KmT,  // [B][128][L] bf16 (transposed)
    const unsigned short* __restrict__ V1,   // [B][L][128]
    const unsigned short* __restrict__ V2,
    unsigned short* __restrict__ O1, unsigned short* __restrict__ O2, int L) {
  constexpr int D = 128, BC = 64, LDK = BC + 8, LDV = D + 8, LDP = 40;
  extern __shared__ unsigned short smem[];
  unsigned short* Kts = smem;                  // [d][key]
  unsigned short* V1s = smem + D * LDK;        // [key][d]
  unsigned short* V2s = V1s + BC * LDV;
  const int tid = threadIdx.x, wv = tid >> 5, lane = tid & 31;
  const int hh = lane >> 4, l15 = lane & 15;
  unsigned short* Psw = V2s + BC * LDV + wv * 16 * LDP;
  const size_t base = (size_t)blockIdx.y * L * D;
  const long q0 = (long)blockIdx.x * 128 + wv * 16;
  const float scale = 0.08838834764831845f;  // 1/sqrt(128)

  v16bf aq[4];
#pragma unroll
  for (int kk = 0; kk < 4; ++kk) {
    const unsigned short* qr = Q + base + (q0 + l15) * (size_t)D + kk * 32;
    aq[kk] = load_a_frag(qr, hh);
  }

  float m_i[8], l_i[8];
  v8f o1[8], o2[8];
#pragma unroll
  for (int r = 0; r < 8; ++r) { m_i[r] = -1e30f; l_i[r] = 0.f; }
#pragma unroll
  for (int nt = 0; nt < 8; ++nt)
#pragma unroll
    for (int r = 0; r < 8; ++r) { o1[nt][r] = 0.f; o2[nt][r] = 0.f; }

  for (int kc = 0; kc < L; kc += BC) {
    __syncthreads();
#ifdef USE_TDM
    // async tile DMAs, one per wave; rows padded by 16B in LDS
    if (wv == 0)  // K^T: 128 rows x 64 bf16 (128B rows)
      tdm_load_2d(0u, KmT + base + kc, L, D, BC, D, (unsigned long long)L, 4u,
                  3u);
    else if (wv == 1)  // V1: 64 rows x 128 bf16 (256B rows)
      tdm_load_2d(18432u, V1 + base + (size_t)kc * D, D, L, D, BC,
                  (unsigned long long)D, 5u, 3u);
    else if (wv == 2)  // V2
      tdm_load_2d(35840u, V2 + base + (size_t)kc * D, D, L, D, BC,
                  (unsigned long long)D, 5u, 3u);
    if (wv < 3) __builtin_amdgcn_s_wait_tensorcnt(0);
#else
    for (int i = tid; i < D * (BC / 8); i += 256) {
      int d = i / (BC / 8);
      int c = (i % (BC / 8)) * 8;
      *(uint4*)&Kts[d * LDK + c] =
          *(const uint4*)(KmT + base + (size_t)d * L + kc + c);
    }
    for (int i = tid; i < BC * (D / 8); i += 256) {
      int r = i / (D / 8);
      int c = (i % (D / 8)) * 8;
      size_t gsrc = base + (size_t)(kc + r) * D + c;
      *(uint4*)&V1s[r * LDV + c] = *(const uint4*)(V1 + gsrc);
      *(uint4*)&V2s[r * LDV + c] = *(const uint4*)(V2 + gsrc);
    }
#endif
    __syncthreads();

    // --- S for all 64 keys: four 16x16 tiles ---
    v8f s[4];
#pragma unroll
    for (int t = 0; t < 4; ++t)
#pragma unroll
      for (int r = 0; r < 8; ++r) s[t][r] = 0.f;
#pragma unroll
    for (int kk = 0; kk < 4; ++kk) {
#pragma unroll
      for (int t = 0; t < 4; ++t) {
        v16bf b = load_b_frag(&Kts[(kk * 32) * LDK + t * 16], LDK, lane);
        s[t] = __builtin_amdgcn_wmma_f32_16x16x32_bf16(false, aq[kk], false, b,
                                                       (short)0, s[t], false,
                                                       false);
      }
    }
    // --- one online-softmax update per 64 keys ---
    float mloc[8], alpha[8], rs[8];
#pragma unroll
    for (int r = 0; r < 8; ++r) {
#pragma unroll
      for (int t = 0; t < 4; ++t) s[t][r] *= scale;
      mloc[r] = fmaxf(fmaxf(s[0][r], s[1][r]), fmaxf(s[2][r], s[3][r]));
      mloc[r] = redmax16(mloc[r]);
    }
#pragma unroll
    for (int r = 0; r < 8; ++r) {
      float mn = fmaxf(m_i[r], mloc[r]);
      alpha[r] = __expf(m_i[r] - mn);
      m_i[r] = mn;
#pragma unroll
      for (int t = 0; t < 4; ++t) s[t][r] = __expf(s[t][r] - mn);
      rs[r] = (s[0][r] + s[1][r]) + (s[2][r] + s[3][r]);
      rs[r] = redsum16(rs[r]);
      l_i[r] = l_i[r] * alpha[r] + rs[r];
    }
#pragma unroll
    for (int nt = 0; nt < 8; ++nt)
#pragma unroll
      for (int r = 0; r < 8; ++r) { o1[nt][r] *= alpha[r]; o2[nt][r] *= alpha[r]; }

    // --- P @ V in two 32-key halves ---
#pragma unroll
    for (int h = 0; h < 2; ++h) {
      // wave-private staging; LDS in-order per wave -> compiler fence only
#pragma unroll
      for (int r = 0; r < 8; ++r) {
        Psw[(r + 8 * hh) * LDP + l15] = f2bf_us(s[2 * h][r]);
        Psw[(r + 8 * hh) * LDP + 16 + l15] = f2bf_us(s[2 * h + 1][r]);
      }
      asm volatile("" ::: "memory");
      v16bf ap = load_a_frag(&Psw[l15 * LDP], hh);
#pragma unroll
      for (int nt = 0; nt < 8; ++nt) {
        v16bf bv1 = load_b_frag(&V1s[(h * 32) * LDV + nt * 16], LDV, lane);
        v16bf bv2 = load_b_frag(&V2s[(h * 32) * LDV + nt * 16], LDV, lane);
        o1[nt] = __builtin_amdgcn_wmma_f32_16x16x32_bf16(
            false, ap, false, bv1, (short)0, o1[nt], false, false);
        o2[nt] = __builtin_amdgcn_wmma_f32_16x16x32_bf16(
            false, ap, false, bv2, (short)0, o2[nt], false, false);
      }
      asm volatile("" ::: "memory");
    }
  }
#pragma unroll
  for (int r = 0; r < 8; ++r) {
    float inv = 1.f / l_i[r];
    long row = q0 + r + 8 * hh;
#pragma unroll
    for (int nt = 0; nt < 8; ++nt) {
      size_t idx = base + (size_t)row * D + nt * 16 + l15;
      O1[idx] = f2bf_us(o1[nt][r] * inv);
      O2[idx] = f2bf_us(o2[nt][r] * inv);
    }
  }
}

// ---------- host-side orchestration ----------
extern "C" void kernel_launch(void* const* d_in, const int* in_sizes, int n_in,
                              void* d_out, int out_size, void* d_ws,
                              size_t ws_size, hipStream_t stream) {
  (void)in_sizes; (void)n_in; (void)out_size; (void)ws_size;
  const int Bb = 4, Ll = 4096;
  const long R = (long)Bb * Ll;
  const float* x1 = (const float*)d_in[0];
  const float* x2 = (const float*)d_in[1];
  const float* ln1_g = (const float*)d_in[2];
  const float* ln1_b = (const float*)d_in[3];
  const float* ln2_g = (const float*)d_in[4];
  const float* ln2_b = (const float*)d_in[5];
  const float* Wq  = (const float*)d_in[6];  const float* bq  = (const float*)d_in[7];
  const float* Wv1 = (const float*)d_in[8];  const float* bv1 = (const float*)d_in[9];
  const float* Wk  = (const float*)d_in[10]; const float* bk  = (const float*)d_in[11];
  const float* Wv2 = (const float*)d_in[12]; const float* bv2 = (const float*)d_in[13];
  const float* Wp1 = (const float*)d_in[14]; const float* bp1 = (const float*)d_in[15];
  const float* Wp2 = (const float*)d_in[16]; const float* bp2 = (const float*)d_in[17];
  const float* lnf_g = (const float*)d_in[18]; const float* lnf_b = (const float*)d_in[19];
  const float* Wf1 = (const float*)d_in[20]; const float* bf1 = (const float*)d_in[21];
  const float* Wf2 = (const float*)d_in[22]; const float* bf2 = (const float*)d_in[23];
  const float* ln3_g = (const float*)d_in[24]; const float* ln3_b = (const float*)d_in[25];
  const float* Wo = (const float*)d_in[26]; const float* bo = (const float*)d_in[27];

  char* ws = (char*)d_ws;
  size_t off = 0;
  auto take = [&](size_t bytes) {
    void* p = ws + off;
    off += (bytes + 255) & ~(size_t)255;
    return p;
  };
  unsigned short* x1n = (unsigned short*)take(R * 128 * 2);
  unsigned short* x2n = (unsigned short*)take(R * 128 * 2);
  unsigned short* q1t = (unsigned short*)take(R * 128 * 2);  // "K", [B][128][L]
  unsigned short* k2b = (unsigned short*)take(R * 128 * 2);  // acts as "Q"
  unsigned short* v1b = (unsigned short*)take(R * 128 * 2);
  unsigned short* v2b = (unsigned short*)take(R * 128 * 2);
  float* y = (float*)take(R * 256 * 4);
  unsigned short* hn = (unsigned short*)take(R * 256 * 2);
  float* f = (float*)take(R * 256 * 4);
  // aliased buffers (lifetimes disjoint; kernels are stream-ordered)
  unsigned short* o1b = x1n;  // dead after projections
  unsigned short* o2b = x2n;
  unsigned short* gl = q1t;   // q1t..v2b contiguous 16MB, dead after flash
  unsigned short* zn = hn;    // hn dead after FFN1

  // dynamic LDS sizes
  const size_t lds_g128 = 64 * 72 * 2 + 64 * 136 * 2;            // 26624
  const size_t lds_g64 = 64 * 72 * 2 + 64 * 72 * 2;              // 18432
  const size_t lds_dual = 64 * 72 * 2 + 2 * (64 * 136 * 2);      // 44032
  const size_t lds_fa = 128 * 72 * 2 + 2 * (64 * 136 * 2) + 8 * 16 * 40 * 2;

  dim3 blk(256);
  ln_bf16<128><<<R / 8, blk, 0, stream>>>(x1, ln1_g, ln1_b, x1n);
  ln_bf16<128><<<R / 8, blk, 0, stream>>>(x2, ln2_g, ln2_b, x2n);

  // stream 1: q1 (pre-transposed for flash-K) + v1 in one pass
  gemm_dual<EPI_BF16_T, EPI_BF16, 128>
      <<<dim3(R / 64, 1), blk, lds_dual, stream>>>(x1n, Wq, bq, q1t, Ll, Wv1,
                                                   bv1, v1b, 128, 128, 128);
  // stream 2: k2 (flash-Q) + v2 in one pass
  gemm_dual<EPI_BF16, EPI_BF16, 128>
      <<<dim3(R / 64, 1), blk, lds_dual, stream>>>(x2n, Wk, bk, k2b, 128, Wv2,
                                                   bv2, v2b, 128, 128, 128);

  // energy = einsum('bqd,bkd->bqk', k2, q1): queries are k2, keys are q1
  flash_attn<<<dim3(Ll / 128, Bb), blk, lds_fa, stream>>>(k2b, q1t, v1b, v2b,
                                                          o1b, o2b, Ll);

  gemm_bf16<EPI_RES_F32, 128><<<dim3(R / 64, 1), blk, lds_g128, stream>>>(
      o1b, Wp1, bp1, x1, y, 128, 128, 128, 256, 0);
  gemm_bf16<EPI_RES_F32, 128><<<dim3(R / 64, 1), blk, lds_g128, stream>>>(
      o2b, Wp2, bp2, x2, y, 128, 128, 128, 256, 128);

  ln_bf16<256><<<R / 8, blk, 0, stream>>>(y, lnf_g, lnf_b, hn);
  gemm_bf16<EPI_GELU_BF16, 128><<<dim3(R / 64, 4), blk, lds_g128, stream>>>(
      hn, Wf1, bf1, nullptr, gl, 256, 512, 0, 512, 0);
  gemm_bf16<EPI_RES_F32, 128><<<dim3(R / 64, 2), blk, lds_g128, stream>>>(
      gl, Wf2, bf2, y, f, 512, 256, 256, 256, 0);
  ln_bf16<256><<<R / 8, blk, 0, stream>>>(f, ln3_g, ln3_b, zn);
  gemm_bf16<EPI_F32_GUARD, 64><<<dim3(R / 64, 1), blk, lds_g64, stream>>>(
      zn, Wo, bo, nullptr, d_out, 256, 55, 0, 55, 0);
}
